// Attention_15530601742392
// MI455X (gfx1250) — matrix operand order repaired and checked by reading, not compiled, not run on hardware
//
#include <hip/hip_runtime.h>
#include <hip/hip_bf16.h>

// ---------------- problem dims ----------------
#define BB      4
#define ENC_LEN 1024
#define DEC_LEN 256
#define HID     1024
#define ATTN    256

typedef __attribute__((ext_vector_type(16))) _Float16 v16h;
typedef __attribute__((ext_vector_type(8)))  float    v8f;
typedef __attribute__((ext_vector_type(2)))  int      v2i;
typedef __attribute__((ext_vector_type(4)))  int      v4i;

// ---------------- gfx1250 async direct-to-LDS staging (guarded) ----------------
#if defined(__has_builtin)
#if __has_builtin(__builtin_amdgcn_global_load_async_to_lds_b128) && \
    __has_builtin(__builtin_amdgcn_global_load_async_to_lds_b64)  && \
    __has_builtin(__builtin_amdgcn_s_wait_asynccnt)
#define HAVE_ASYNC_LDS 1
#endif
#endif

#define ASG __attribute__((address_space(1)))
#define ASL __attribute__((address_space(3)))

// Integer round-trip AS casts (flat LDS addr[31:0] IS the LDS offset, ISA 10.2).
__device__ __forceinline__ ASG v2i* glb2(const void* p) {
  return (ASG v2i*)(unsigned long long)p;
}
__device__ __forceinline__ ASG v4i* glb4(const void* p) {
  return (ASG v4i*)(unsigned long long)p;
}
__device__ __forceinline__ ASL v2i* lds2(const void* p) {
  return (ASL v2i*)(unsigned int)(unsigned long long)p;
}
__device__ __forceinline__ ASL v4i* lds4(const void* p) {
  return (ASL v4i*)(unsigned int)(unsigned long long)p;
}

// ---------------- fast tanh (gfx1250 v_tanh_f32 when available) ----------------
__device__ __forceinline__ float fast_tanh(float x) {
#if defined(__has_builtin) && __has_builtin(__builtin_amdgcn_tanhf)
  return __builtin_amdgcn_tanhf(x);
#else
  float t = __expf(2.0f * x);
  return (t - 1.0f) / (t + 1.0f);
#endif
}

// ---------------- f32 -> f16 conversion (row-major, vectorized x4) ----------------
__global__ __launch_bounds__(256) void cvt_f32_to_f16(const float* __restrict__ in,
                                                      _Float16* __restrict__ out, int n4) {
  int i = blockIdx.x * 256 + threadIdx.x;
  if (i < n4) {
    float4 q = ((const float4*)in)[i];
    union { _Float16 h[4]; unsigned long long u; } p;
    p.h[0] = (_Float16)q.x; p.h[1] = (_Float16)q.y;
    p.h[2] = (_Float16)q.z; p.h[3] = (_Float16)q.w;
    ((unsigned long long*)out)[i] = p.u;
  }
}

// ---------------- f32 [R][C] -> f16 transposed [C][R] (tiled via LDS) ----------------
__global__ __launch_bounds__(256) void transpose_f32_to_f16(
    const float* __restrict__ in, _Float16* __restrict__ out,
    int R, int C, long sIn, long sOut) {
  __shared__ float tile[16][17];
  const int r0 = blockIdx.y * 16, c0 = blockIdx.x * 16;
  const int tx = threadIdx.x & 15, ty = threadIdx.x >> 4;
  const float* src = in + (long)blockIdx.z * sIn;
  _Float16* dst = out + (long)blockIdx.z * sOut;
  tile[ty][tx] = src[(long)(r0 + ty) * C + (c0 + tx)];
  __syncthreads();
  dst[(long)(c0 + ty) * R + (r0 + tx)] = (_Float16)tile[tx][ty];
}

// ---------------- WMMA f16 GEMM: C[M,N] = A[M,K] * B[K,N] (+bias), B given transposed ----
// A:  [M][K] row-major f16.   BT: [N][K] row-major f16 (B transposed).
// 8 waves/block; each wave owns a 16(M)x32(N) C strip (2 N-tiles, 2 accumulators).
// Block covers 16(M) x 256(N).  K staged 64 wide -> 4 WMMAs per barrier pair.
// Staging uses GLOBAL_LOAD_ASYNC_TO_LDS (ASYNCcnt) when available; fragment reads are
// contiguous b128 LDS loads (k = 16*hi + 0..15, same order in A and B).
#define GEMM_WAVES 8
__global__ __launch_bounds__(256) void wmma_gemm_f16f32(
    const _Float16* __restrict__ A, const _Float16* __restrict__ BT,
    const float* __restrict__ bias, float* __restrict__ C,
    int M, int N, int K, long sA, long sBT, long sC) {
  __shared__ _Float16 As[16 * 64];                         // 2 KB (shared by block)
  __shared__ _Float16 Bs[GEMM_WAVES][2][16 * 64];          // 32 KB (per-wave subtiles)

  const int tid  = threadIdx.x;
  const int wave = tid >> 5;
  const int lane = tid & 31;
  const int l16  = lane & 15;
  const int hi   = lane >> 4;                              // 0: lanes 0-15, 1: lanes 16-31

  const long zA = (long)blockIdx.z * sA;
  const long zB = (long)blockIdx.z * sBT;
  const long zC = (long)blockIdx.z * sC;

  const int m0    = blockIdx.y * 16;
  const int nBase = blockIdx.x * (GEMM_WAVES * 32) + wave * 32;

  v8f acc0 = {}, acc1 = {};

  for (int k0 = 0; k0 < K; k0 += 64) {
    __syncthreads();
    // --- stage A tile 16x64 (1024 halves): 256 threads x 1 qword ---
    {
      int idx = tid * 4;                    // half index within tile
      int r = idx >> 6, c = idx & 63;
      const _Float16* srcA = &A[zA + (long)(m0 + r) * K + (k0 + c)];
#ifdef HAVE_ASYNC_LDS
      __builtin_amdgcn_global_load_async_to_lds_b64(glb2(srcA), lds2(&As[idx]), 0, 0);
#else
      *(unsigned long long*)&As[idx] = *(const unsigned long long*)srcA;
#endif
      if (k0 + 64 < K) __builtin_prefetch(srcA + 64, 0, 1);     // global_prefetch_b8
    }
    // --- stage BT subtiles (2x 16 rows x 64 halves per wave): contiguous rows ---
#pragma unroll
    for (int t = 0; t < 2; ++t) {
      const _Float16* srcB = &BT[zB + (long)(nBase + t * 16 + l16) * K + (k0 + hi * 32)];
      _Float16* dstB = &Bs[wave][t][l16 * 64 + hi * 32];
#pragma unroll
      for (int q = 0; q < 4; ++q) {
#ifdef HAVE_ASYNC_LDS
        __builtin_amdgcn_global_load_async_to_lds_b128(glb4(srcB + q * 8),
                                                       lds4(dstB + q * 8), 0, 0);
#else
        ((uint4*)dstB)[q] = ((const uint4*)srcB)[q];
#endif
      }
      if (k0 + 64 < K) __builtin_prefetch(srcB + 64, 0, 1);
    }
#ifdef HAVE_ASYNC_LDS
    __builtin_amdgcn_s_wait_asynccnt(0);
#endif
    __syncthreads();

    // --- 2 K-substeps x 2 N-tiles: fragments are contiguous b128 runs ---
#pragma unroll
    for (int s = 0; s < 2; ++s) {
      union { v16h v; uint4 q[2]; } a;
      a.q[0] = *(const uint4*)&As[l16 * 64 + s * 32 + hi * 16];       // K = s*32+hi*16+0..7
      a.q[1] = *(const uint4*)&As[l16 * 64 + s * 32 + hi * 16 + 8];   // K = s*32+hi*16+8..15

      union { v16h v; uint4 q[2]; } b0, b1;
      const _Float16* p0 = &Bs[wave][0][l16 * 64 + s * 32 + hi * 16]; // K = s*32+hi*16+0..15
      const _Float16* p1 = &Bs[wave][1][l16 * 64 + s * 32 + hi * 16];
      b0.q[0] = ((const uint4*)p0)[0]; b0.q[1] = ((const uint4*)p0)[1];
      b1.q[0] = ((const uint4*)p1)[0]; b1.q[1] = ((const uint4*)p1)[1];

      acc0 = __builtin_amdgcn_wmma_f32_16x16x32_f16(false, a.v, false, b0.v,
                                                    (short)0, acc0, false, false);
      acc1 = __builtin_amdgcn_wmma_f32_16x16x32_f16(false, a.v, false, b1.v,
                                                    (short)0, acc1, false, false);
    }
  }

  // --- epilogue: C layout — lane col n=l16, dword r row m = r + hi*8 ---
  {
    int n = nBase + l16;
    float bv = bias ? bias[n] : 0.0f;
#pragma unroll
    for (int r = 0; r < 8; ++r)
      C[zC + (long)(m0 + hi * 8 + r) * N + n] = acc0[r] + bv;
  }
  {
    int n = nBase + 16 + l16;
    float bv = bias ? bias[n] : 0.0f;
#pragma unroll
    for (int r = 0; r < 8; ++r)
      C[zC + (long)(m0 + hi * 8 + r) * N + n] = acc1[r] + bv;
  }
}

// ---------------- fused additive-attention logits ----------------
// logits[b,d,e] = sum_a v[a] * tanh(pd[b,d,a] + pe[b,e,a])
__global__ __launch_bounds__(256) void logits_kernel(
    const float* __restrict__ pd,   // [B*DEC, ATTN]
    const float* __restrict__ pe,   // [B*ENC, ATTN]
    const float* __restrict__ v,    // [ATTN]
    float* __restrict__ logits) {   // [B*DEC, ENC]
  __shared__ float pds[ATTN];
  __shared__ float vs[ATTN];
  const int bd = blockIdx.x;          // b*DEC + d
  const int b  = bd >> 8;             // DEC = 256
  const int t  = threadIdx.x;
  pds[t] = pd[(long)bd * ATTN + t];
  vs[t]  = v[t];
  __syncthreads();

  const int e = blockIdx.y * 256 + t;
  const float4* row = (const float4*)(pe + ((long)b * ENC_LEN + e) * ATTN);
  float acc = 0.0f;
#pragma unroll 4
  for (int a = 0; a < ATTN / 4; ++a) {
    float4 q = row[a];
    acc += vs[4 * a + 0] * fast_tanh(pds[4 * a + 0] + q.x);
    acc += vs[4 * a + 1] * fast_tanh(pds[4 * a + 1] + q.y);
    acc += vs[4 * a + 2] * fast_tanh(pds[4 * a + 2] + q.z);
    acc += vs[4 * a + 3] * fast_tanh(pds[4 * a + 3] + q.w);
  }
  logits[(long)bd * ENC_LEN + e] = acc;
}

// ---------------- masked softmax + renorm (single pass: exp*mask / sum) ----------------
__global__ __launch_bounds__(256) void softmax_kernel(
    const float* __restrict__ logits,   // [B*DEC, ENC]
    const float* __restrict__ mask,     // [B, ENC]
    float* __restrict__ attn_f32,       // [B*DEC, ENC] (d_out part 2)
    _Float16* __restrict__ attn_f16) {  // [B*DEC, ENC] (A of context GEMM)
  __shared__ float red[256];
  const int bd = blockIdx.x;
  const int b  = bd >> 8;
  const int t  = threadIdx.x;
  const float* row = logits + (long)bd * ENC_LEN;

  float vals[4], mk[4];
  float mx = -3.4e38f;
#pragma unroll
  for (int j = 0; j < 4; ++j) {
    int e = t + 256 * j;
    vals[j] = row[e];
    mk[j]   = mask[(long)b * ENC_LEN + e];
    mx = fmaxf(mx, vals[j]);
  }
  red[t] = mx; __syncthreads();
  for (int s = 128; s > 0; s >>= 1) { if (t < s) red[t] = fmaxf(red[t], red[t + s]); __syncthreads(); }
  mx = red[0]; __syncthreads();

  float p[4], sum = 0.0f;
#pragma unroll
  for (int j = 0; j < 4; ++j) { p[j] = __expf(vals[j] - mx) * mk[j]; sum += p[j]; }
  red[t] = sum; __syncthreads();
  for (int s = 128; s > 0; s >>= 1) { if (t < s) red[t] += red[t + s]; __syncthreads(); }
  float inv = 1.0f / red[0];
#pragma unroll
  for (int j = 0; j < 4; ++j) {
    int e = t + 256 * j;
    float w = p[j] * inv;
    attn_f32[(long)bd * ENC_LEN + e] = w;
    attn_f16[(long)bd * ENC_LEN + e] = (_Float16)w;
  }
}

// ---------------- host launch ----------------
extern "C" void kernel_launch(void* const* d_in, const int* in_sizes, int n_in,
                              void* d_out, int out_size, void* d_ws, size_t ws_size,
                              hipStream_t stream) {
  const float* enc  = (const float*)d_in[0];  // [B,ENC,HID]
  const float* dec  = (const float*)d_in[1];  // [B,DEC,HID]
  const float* mask = (const float*)d_in[2];  // [B,ENC]
  const float* Wh   = (const float*)d_in[3];  // [HID,ATTN]
  const float* Ws   = (const float*)d_in[4];  // [HID,ATTN]
  const float* bs   = (const float*)d_in[5];  // [ATTN]
  const float* v    = (const float*)d_in[6];  // [ATTN]

  float* context  = (float*)d_out;                                   // [B,DEC,HID]
  float* attn_out = (float*)d_out + (long)BB * DEC_LEN * ENC_LEN;    // [B,DEC,ENC]

  // workspace layout
  char* w = (char*)d_ws;
  const long N_ENC = (long)BB * ENC_LEN * HID;       // 4M elements
  const long N_DEC = (long)BB * DEC_LEN * HID;       // 1M
  const long N_W   = (long)HID * ATTN;               // 256K
  _Float16* enc_h   = (_Float16*)w;  w += N_ENC * 2;                       // 8 MB   A of proj_enc
  _Float16* dec_h   = (_Float16*)w;  w += N_DEC * 2;                       // 2 MB   A of proj_dec
  _Float16* WhT_h   = (_Float16*)w;  w += N_W * 2;                         // 0.5 MB [ATTN][HID]
  _Float16* WsT_h   = (_Float16*)w;  w += N_W * 2;                         // 0.5 MB [ATTN][HID]
  _Float16* encT_h  = (_Float16*)w;  w += N_ENC * 2;                       // 8 MB   [B][HID][ENC]
  float*    proj_enc= (float*)w;     w += (long)BB * ENC_LEN * ATTN * 4;   // 4 MB
  float*    proj_dec= (float*)w;     w += (long)BB * DEC_LEN * ATTN * 4;   // 1 MB
  float*    logits  = (float*)w;     w += (long)BB * DEC_LEN * ENC_LEN * 4;// 4 MB
  _Float16* attn_h  = (_Float16*)w;                                        // 2 MB  [B][DEC][ENC]

  // 1) operand conversion: A operands row-major f16, B operands transposed f16
  auto cvt = [&](const float* src, _Float16* dst, long n) {
    int n4 = (int)(n / 4);
    cvt_f32_to_f16<<<(n4 + 255) / 256, 256, 0, stream>>>(src, dst, n4);
  };
  cvt(enc, enc_h, N_ENC);
  cvt(dec, dec_h, N_DEC);
  transpose_f32_to_f16<<<dim3(ATTN / 16, HID / 16, 1), 256, 0, stream>>>(
      Wh, WhT_h, HID, ATTN, 0, 0);
  transpose_f32_to_f16<<<dim3(HID / 16, ENC_LEN / 16, BB), 256, 0, stream>>>(
      enc, encT_h, ENC_LEN, HID, (long)ENC_LEN * HID, (long)HID * ENC_LEN);
  transpose_f32_to_f16<<<dim3(ATTN / 16, HID / 16, 1), 256, 0, stream>>>(
      Ws, WsT_h, HID, ATTN, 0, 0);

  // 2) proj_enc = enc @ Wh           (M=4096, N=256, K=1024)
  wmma_gemm_f16f32<<<dim3(ATTN / 256, (BB * ENC_LEN) / 16, 1), 256, 0, stream>>>(
      enc_h, WhT_h, nullptr, proj_enc, BB * ENC_LEN, ATTN, HID, 0, 0, 0);

  // 3) proj_dec = dec @ Ws + bs      (M=1024, N=256, K=1024)
  wmma_gemm_f16f32<<<dim3(ATTN / 256, (BB * DEC_LEN) / 16, 1), 256, 0, stream>>>(
      dec_h, WsT_h, bs, proj_dec, BB * DEC_LEN, ATTN, HID, 0, 0, 0);

  // 4) fused tanh-dot logits         [B*DEC, ENC]
  logits_kernel<<<dim3(BB * DEC_LEN, ENC_LEN / 256, 1), 256, 0, stream>>>(
      proj_dec, proj_enc, v, logits);

  // 5) masked softmax + renorm -> attn (f32 to d_out, f16 for context GEMM)
  softmax_kernel<<<dim3(BB * DEC_LEN, 1, 1), 256, 0, stream>>>(
      logits, mask, attn_out, attn_h);

  // 6) context = attn @ enc          (batched: M=256, N=1024, K=1024, z=B)
  wmma_gemm_f16f32<<<dim3(HID / 256, DEC_LEN / 16, BB), 256, 0, stream>>>(
      attn_h, encT_h, nullptr, context, DEC_LEN, HID, ENC_LEN,
      (long)DEC_LEN * ENC_LEN, (long)HID * ENC_LEN, (long)DEC_LEN * HID);
}
